// TokenLevelPruner_81149112090678
// MI455X (gfx1250) — matrix operand order repaired and checked by reading, not compiled, not run on hardware
//
#include <hip/hip_runtime.h>
#include <hip/hip_bf16.h>
#include <math.h>

#define B_SZ    16
#define N_TOK   16384
#define C_DIM   256
#define HID     128
#define KKEEP   11468                    // int(16384 * 0.7)
#define N_TILES (N_TOK / 16)             // 1024 M-tiles per batch
#define TILES_TOTAL (B_SZ * N_TILES)     // 16384 waves

#define PRUNED_ELEMS ((size_t)B_SZ * KKEEP * C_DIM)
#define IDX_OFF      PRUNED_ELEMS
#define IDX_ELEMS    ((size_t)B_SZ * KKEEP)
#define SCORES_OFF   (IDX_OFF + IDX_ELEMS)

typedef __attribute__((ext_vector_type(16))) __bf16 v16bf;
typedef __attribute__((ext_vector_type(8)))  float  v8f;

// Monotone float -> uint key (ascending key == ascending float)
__device__ __forceinline__ unsigned int score_key(float f) {
    unsigned int u = __float_as_uint(f);
    unsigned int m = (unsigned int)((int)u >> 31) | 0x80000000u;
    return u ^ m;
}

// ---------------------------------------------------------------------------
// Stage w1 [C=256(K) x HID=128(N)] fp32 -> bf16 in WMMA B-matrix layout.
// Staged layout: block (kb, nt) of 32Kx16N; per lane 16 contiguous bf16:
//   lane l: N = nt*16 + (l&15), K = kb*32 + (l>>4)*8 + (j&7) + (j>>3)*16, j = 0..15
// ---------------------------------------------------------------------------
__global__ __launch_bounds__(256) void stage_w1_kernel(
    const float* __restrict__ w1, __bf16* __restrict__ w1s) {
    int i = blockIdx.x * 256 + threadIdx.x;      // [0, 256*128)
    int j    = i & 15;
    int lane = (i >> 4) & 31;
    int tile = i >> 9;                           // kb*8 + nt
    int kb = tile >> 3, nt = tile & 7;
    int k = kb * 32 + (lane >> 4) * 8 + (j & 7) + (j >> 3) * 16;
    int n = nt * 16 + (lane & 15);
    w1s[i] = (__bf16)w1[k * HID + n];
}

// ---------------------------------------------------------------------------
// Fused scorer: h = GELU(tokens@w1 + b1); scores = h@w2 + b2
// One wave per 16-token tile; 64 x v_wmma_f32_16x16x32_bf16 per wave.
// ---------------------------------------------------------------------------
__global__ __launch_bounds__(256) void scorer_kernel(
    const float* __restrict__ tokens,
    const float* __restrict__ b1,
    const float* __restrict__ w2,
    const float* __restrict__ b2,
    const __bf16* __restrict__ w1s,
    float* __restrict__ scores) {
    int wave = (blockIdx.x * 256 + threadIdx.x) >> 5;
    int lane = threadIdx.x & 31;
    if (wave >= TILES_TOTAL) return;
    int b    = wave >> 10;                 // / N_TILES
    int tn   = wave & (N_TILES - 1);
    int half = lane >> 4;
    int lmod = lane & 15;

    const float* Abase = tokens + (((size_t)b * N_TOK) + (size_t)tn * 16 + lmod) * C_DIM;
    const v16bf* Bv = (const v16bf*)w1s;

    v8f acc[8];
    const v8f zero = {0.f, 0.f, 0.f, 0.f, 0.f, 0.f, 0.f, 0.f};
#pragma unroll
    for (int nt = 0; nt < 8; ++nt) acc[nt] = zero;

#pragma unroll
    for (int kb = 0; kb < 8; ++kb) {
        // A-matrix 16-bit 16x32 layout: lane half h holds K = kb*32 + h*8 + [0..7]
        // in elems 0..7 and K = +16 in elems 8..15.
        const float* ap = Abase + kb * 32 + half * 8;
        float4 f0 = *(const float4*)(ap + 0);
        float4 f1 = *(const float4*)(ap + 4);
        float4 f2 = *(const float4*)(ap + 16);
        float4 f3 = *(const float4*)(ap + 20);
        v16bf a;
        a[0]  = (__bf16)f0.x; a[1]  = (__bf16)f0.y; a[2]  = (__bf16)f0.z; a[3]  = (__bf16)f0.w;
        a[4]  = (__bf16)f1.x; a[5]  = (__bf16)f1.y; a[6]  = (__bf16)f1.z; a[7]  = (__bf16)f1.w;
        a[8]  = (__bf16)f2.x; a[9]  = (__bf16)f2.y; a[10] = (__bf16)f2.z; a[11] = (__bf16)f2.w;
        a[12] = (__bf16)f3.x; a[13] = (__bf16)f3.y; a[14] = (__bf16)f3.z; a[15] = (__bf16)f3.w;
#pragma unroll
        for (int nt = 0; nt < 8; ++nt) {
            v16bf bv = Bv[((kb * 8 + nt) << 5) + lane];
            acc[nt] = __builtin_amdgcn_wmma_f32_16x16x32_bf16(
                false, a, false, bv, (short)0, acc[nt], false, false);
        }
    }

    // Epilogue: C/D layout => VGPR v, lane: M = v + half*8, N = nt*16 + lmod.
    // score[M] = sum_N gelu(h + b1[N]) * w2[N]  (+ b2)
    float sc[8];
#pragma unroll
    for (int v = 0; v < 8; ++v) {
        float s = 0.f;
#pragma unroll
        for (int nt = 0; nt < 8; ++nt) {
            int n = nt * 16 + lmod;
            float x = acc[nt][v] + b1[n];
            float g = 0.5f * x * (1.0f + erff(x * 0.70710678118654752f));
            s += g * w2[n];
        }
        // reduce across the 16 lanes of each half-wave
        s += __shfl_xor(s, 1, 32);
        s += __shfl_xor(s, 2, 32);
        s += __shfl_xor(s, 4, 32);
        s += __shfl_xor(s, 8, 32);
        sc[v] = s;
    }
    if (lmod == 0) {
        float bias = b2[0];
#pragma unroll
        for (int v = 0; v < 8; ++v) {
            scores[(size_t)b * N_TOK + tn * 16 + half * 8 + v] = sc[v] + bias;
        }
    }
}

// ---------------------------------------------------------------------------
// Exact top-K threshold via 4-pass MSB radix select. One block per batch.
// Output per batch: (threshold key, #ties-to-keep). count(key>T) = K - ties.
// ---------------------------------------------------------------------------
__global__ __launch_bounds__(256) void select_kernel(
    const float* __restrict__ scores, uint2* __restrict__ thresh) {
    __shared__ unsigned int hist[256];
    __shared__ unsigned int sh_prefix, sh_k;
    int b = blockIdx.x;
    const float* s = scores + (size_t)b * N_TOK;

    unsigned int prefix = 0;
    unsigned int kk = KKEEP;
    for (int d = 24; d >= 0; d -= 8) {
        hist[threadIdx.x] = 0;
        __syncthreads();
        unsigned int mask = (d == 24) ? 0u : (0xFFFFFFFFu << (d + 8));
        for (int i = threadIdx.x; i < N_TOK; i += 256) {
            unsigned int key = score_key(s[i]);
            if ((key & mask) == prefix)
                atomicAdd(&hist[(key >> d) & 255], 1u);
        }
        __syncthreads();
        if (threadIdx.x == 0) {
            unsigned int k = kk, pb = 0;
            for (int bin = 255; bin >= 0; --bin) {
                unsigned int c = hist[bin];
                if (k <= c) { pb = (unsigned int)bin; break; }
                k -= c;
            }
            sh_prefix = prefix | (pb << d);
            sh_k = k;
        }
        __syncthreads();
        prefix = sh_prefix;
        kk = sh_k;
    }
    if (threadIdx.x == 0) thresh[b] = make_uint2(prefix, kk);
}

// ---------------------------------------------------------------------------
// Compaction: keep (key > T) plus first `ties` elements with key == T (by
// ascending index -> matches lax.top_k tie-break, output sorted for free).
// One 1024-thread block per batch; each thread owns 16 contiguous indices.
// ---------------------------------------------------------------------------
__global__ __launch_bounds__(1024) void compact_kernel(
    const float* __restrict__ scores, const uint2* __restrict__ thresh,
    float* __restrict__ kept_idx) {
    __shared__ unsigned int sdata[1024];
    int b = blockIdx.x;
    unsigned int T    = thresh[b].x;
    unsigned int ties = thresh[b].y;
    int t = threadIdx.x;
    int base = t * 16;
    const float* s = scores + (size_t)b * N_TOK;

    unsigned int keys[16];
    unsigned int gt = 0, eq = 0;
#pragma unroll
    for (int i = 0; i < 16; ++i) {
        keys[i] = score_key(s[base + i]);
        gt += (keys[i] > T);
        eq += (keys[i] == T);
    }
    // block-wide exclusive scan of packed (gt | eq<<16)
    unsigned int v = gt | (eq << 16);
    sdata[t] = v;
    __syncthreads();
    for (int off = 1; off < 1024; off <<= 1) {
        unsigned int x = (t >= off) ? sdata[t - off] : 0u;
        __syncthreads();
        sdata[t] += x;
        __syncthreads();
    }
    unsigned int ex = sdata[t] - v;
    unsigned int gt_before = ex & 0xFFFFu;
    unsigned int eq_before = ex >> 16;

    float* kb = kept_idx + (size_t)b * KKEEP;
#pragma unroll
    for (int i = 0; i < 16; ++i) {
        unsigned int eq_kept = (eq_before < ties) ? eq_before : ties;
        unsigned int pos = gt_before + eq_kept;
        bool keep = (keys[i] > T) || ((keys[i] == T) && (eq_before < ties));
        if (keep) kb[pos] = (float)(base + i);
        gt_before += (keys[i] > T);
        eq_before += (keys[i] == T);
    }
}

// ---------------------------------------------------------------------------
// Gather kept tokens: 64 threads per token, float4 (1 KB / token).
// ---------------------------------------------------------------------------
__global__ __launch_bounds__(256) void gather_kernel(
    const float* __restrict__ tokens, const float* __restrict__ kept_idx,
    float* __restrict__ pruned) {
    int g = blockIdx.x * 4 + (threadIdx.x >> 6);   // token slot in [0, B*KKEEP)
    int lane64 = threadIdx.x & 63;
    int b = g / KKEEP;
    int idx = (int)(kept_idx[g] + 0.5f);
    const float4* src = (const float4*)(tokens + ((size_t)b * N_TOK + idx) * C_DIM);
    float4* dst = (float4*)(pruned + (size_t)g * C_DIM);
    dst[lane64] = src[lane64];
}

// ---------------------------------------------------------------------------
extern "C" void kernel_launch(void* const* d_in, const int* in_sizes, int n_in,
                              void* d_out, int out_size, void* d_ws, size_t ws_size,
                              hipStream_t stream) {
    const float* tokens = (const float*)d_in[0];
    const float* w1     = (const float*)d_in[1];
    const float* b1     = (const float*)d_in[2];
    const float* w2     = (const float*)d_in[3];
    const float* b2     = (const float*)d_in[4];

    float* out      = (float*)d_out;
    float* pruned   = out;                 // [B, K, C]
    float* kept_idx = out + IDX_OFF;       // [B, K] (indices as float)
    float* scores   = out + SCORES_OFF;    // [B, N]

    __bf16* w1s    = (__bf16*)d_ws;                       // 64 KB staged B-matrix
    uint2*  thresh = (uint2*)((char*)d_ws + 65536);       // 16 x {key, ties}

    stage_w1_kernel<<<(C_DIM * HID) / 256, 256, 0, stream>>>(w1, w1s);
    scorer_kernel<<<TILES_TOTAL / 8, 256, 0, stream>>>(tokens, b1, w2, b2, w1s, scores);
    select_kernel<<<B_SZ, 256, 0, stream>>>(scores, thresh);
    compact_kernel<<<B_SZ, 1024, 0, stream>>>(scores, thresh, kept_idx);
    gather_kernel<<<(B_SZ * KKEEP) / 4, 256, 0, stream>>>(tokens, kept_idx, pruned);
}
